// SAGE_12781822673112
// MI455X (gfx1250) — compile-verified, matrix operand order checked by reading, not run on hardware
//
#include <hip/hip_runtime.h>
#include <hip/hip_bf16.h>
#include <stdint.h>

#define N_NODES 100000
#define N_EDGES 1600000
#define IN_CH   128
#define HID_CH  128
#define OUT_CH  64

typedef __attribute__((ext_vector_type(16))) __bf16 v16bf;
typedef __attribute__((ext_vector_type(8)))  float  v8f;

// float -> bf16 bits, round-to-nearest-even (used only in prep passes)
__device__ __forceinline__ unsigned short f2bf(float f) {
    unsigned int x = __float_as_uint(f);
    unsigned int r = x + 0x7FFFu + ((x >> 16) & 1u);
    return (unsigned short)(r >> 16);
}

// Fragment K index for element j of a lane (ISA 16-bit A/B 16x32 layout):
// lane<16 covers K {0..7, 16..23}, lane>=16 covers K {8..15, 24..31}.
__device__ __forceinline__ int frag_kl(int lane, int j) {
    return ((j >> 3) << 4) + (((lane >> 4) & 1) << 3) + (j & 7);
}

// ---------------------------------------------------------------------------
// Zero-fill (graph-capture safe; no hipMemset)
// ---------------------------------------------------------------------------
__global__ void k_fill0(float* __restrict__ p, size_t n) {
    size_t i = (size_t)blockIdx.x * blockDim.x + threadIdx.x;
    size_t stride = (size_t)gridDim.x * blockDim.x;
    for (; i < n; i += stride) p[i] = 0.0f;
}

// ---------------------------------------------------------------------------
// Degree counts: cnt[dst] += 1
// ---------------------------------------------------------------------------
__global__ void k_count(const long long* __restrict__ dst,
                        float* __restrict__ cnt, int E) {
    int e = blockIdx.x * blockDim.x + threadIdx.x;
    if (e < E) atomicAdd(&cnt[(int)dst[e]], 1.0f);
}

// ---------------------------------------------------------------------------
// Edge scatter: agg[dst][c] += feat[src][c].  One thread per (edge, channel),
// C == 128; blockDim = 256 => 2 edges/block.  Gather reads coalesced
// (512B/row); feature + agg buffers are L2-resident (51MB each << 192MB L2),
// so the 204.8M atomic adds resolve in L2.  Lane 0 of each row issues a
// prefetch for a row ~4096 edges ahead to hide gather latency.
// ---------------------------------------------------------------------------
__global__ void k_scatter(const float* __restrict__ feat,
                          const long long* __restrict__ src,
                          const long long* __restrict__ dst,
                          float* __restrict__ agg, int E) {
    const int C = 128;
    int c = threadIdx.x & (C - 1);
    int e = blockIdx.x * (blockDim.x / C) + (threadIdx.x / C);
    if (e >= E) return;
    if (c == 0) {
        int ep = e + 4096;
        if (ep < E) __builtin_prefetch(feat + (size_t)src[ep] * C, 0, 0);
    }
    int s = (int)src[e];
    int d = (int)dst[e];
    float v = feat[(size_t)s * C + c];
    atomicAdd(agg + (size_t)d * C + c, v);
}

// ---------------------------------------------------------------------------
// A-operand prep: fuse mean = agg/max(deg,1) with f32->bf16 conversion and
// write [mean | x] (K = 2*Cin per node) directly in WMMA A-fragment order:
//   Afrag[tile][t][lane][j],  tile = 16 nodes, t = K/32 steps.
// ---------------------------------------------------------------------------
__global__ void k_aprep(const float* __restrict__ agg,
                        const float* __restrict__ xin,
                        const float* __restrict__ cnt,
                        unsigned short* __restrict__ Afrag,
                        int Cin) {
    const int T = (2 * Cin) / 32;
    size_t id = (size_t)blockIdx.x * blockDim.x + threadIdx.x;
    size_t total = (size_t)(N_NODES / 16) * T * 32 * 16;
    if (id >= total) return;

    int j    = (int)(id & 15);
    int lane = (int)((id >> 4) & 31);
    int t    = (int)((id >> 9) % T);
    int tile = (int)((id >> 9) / T);

    int m = (tile << 4) + (lane & 15);
    int k = t * 32 + frag_kl(lane, j);

    float v;
    if (k < Cin) {
        float invc = 1.0f / fmaxf(cnt[m], 1.0f);
        v = agg[(size_t)m * Cin + k] * invc;
    } else {
        v = xin[(size_t)m * Cin + (k - Cin)];
    }
    Afrag[id] = f2bf(v);
}

// ---------------------------------------------------------------------------
// Weight pack: [W_l ; W_r] (2*Cin x Cout, row-major K x N) -> bf16 in
// B-fragment order: Wfrag[ot][t][lane][j], ot = Cout/16 tiles.
// ---------------------------------------------------------------------------
__global__ void k_packw(const float* __restrict__ Wl,
                        const float* __restrict__ Wr,
                        unsigned short* __restrict__ Wfrag,
                        int Cin, int Cout) {
    const int T = (2 * Cin) / 32;
    int id = blockIdx.x * blockDim.x + threadIdx.x;
    int total = (Cout / 16) * T * 32 * 16;
    if (id >= total) return;

    int j    = id & 15;
    int lane = (id >> 4) & 31;
    int t    = (id >> 9) % T;
    int ot   = (id >> 9) / T;

    int n = (ot << 4) + (lane & 15);
    int k = t * 32 + frag_kl(lane, j);

    float v = (k < Cin) ? Wl[k * Cout + n] : Wr[(k - Cin) * Cout + n];
    Wfrag[id] = f2bf(v);
}

// ---------------------------------------------------------------------------
// Fused SAGE combine GEMM:  out = Afrag @ Wfrag + bias  (optional ReLU).
// One 16-node tile per block; one wave per 16-col output tile.  Fragments
// are pre-laid-out, so each K-step is two b128 loads for A, two for B
// (lane-contiguous, fully coalesced), and one v_wmma_f32_16x16x32_bf16.
// EXEC stays all-ones (N, K, Cout all multiples of the tile sizes).
// ---------------------------------------------------------------------------
template <int CIN, int COUT, int RELU>
__global__ void k_sage_gemm(const unsigned short* __restrict__ Afrag,
                            const unsigned short* __restrict__ Wfrag,
                            const float* __restrict__ bias,
                            float* __restrict__ out) {
    constexpr int T = (2 * CIN) / 32;
    const int tile = blockIdx.x;
    const int lane = threadIdx.x & 31;
    const int ot   = threadIdx.x >> 5;
    const int sel8 = ((lane >> 4) & 1) << 3;
    const int n    = (ot << 4) + (lane & 15);

    const v16bf* ap = (const v16bf*)Afrag + (size_t)tile * T * 32 + lane;
    const v16bf* bp = (const v16bf*)Wfrag + (size_t)ot   * T * 32 + lane;

    v8f acc = {};
#pragma unroll
    for (int t = 0; t < T; ++t) {
        v16bf a = ap[t * 32];
        v16bf b = bp[t * 32];
        acc = __builtin_amdgcn_wmma_f32_16x16x32_bf16(
            /*neg_a=*/false, a, /*neg_b=*/false, b,
            /*c_mod=*/(short)0, acc, /*reuse_a=*/false, /*reuse_b=*/false);
    }

    const float bn = bias[n];
    float* orow = out + (size_t)((tile << 4) + sel8) * COUT + n;
#pragma unroll
    for (int r = 0; r < 8; ++r) {
        float v = acc[r] + bn;
        if (RELU) v = fmaxf(v, 0.0f);
        orow[(size_t)r * COUT] = v;
    }
}

// ---------------------------------------------------------------------------
// Launch
// ---------------------------------------------------------------------------
extern "C" void kernel_launch(void* const* d_in, const int* in_sizes, int n_in,
                              void* d_out, int out_size, void* d_ws, size_t ws_size,
                              hipStream_t stream) {
    const float*     x   = (const float*)d_in[0];
    const long long* ei  = (const long long*)d_in[1];   // int64 (2, E)
    const float*     Wl1 = (const float*)d_in[2];
    const float*     Wr1 = (const float*)d_in[3];
    const float*     b1  = (const float*)d_in[4];
    const float*     Wl2 = (const float*)d_in[5];
    const float*     Wr2 = (const float*)d_in[6];
    const float*     b2  = (const float*)d_in[7];
    float*           out = (float*)d_out;

    const long long* src = ei;
    const long long* dst = ei + N_EDGES;

    // Workspace carve-up (256B aligned regions)
    char* ws = (char*)d_ws;
    auto carve = [&](size_t bytes) {
        char* p = ws;
        ws += (bytes + 255) & ~(size_t)255;
        return p;
    };
    float*          cnt   = (float*)         carve((size_t)N_NODES * 4);
    float*          agg   = (float*)         carve((size_t)N_NODES * 128 * 4);  // reused L1/L2
    float*          h     = (float*)         carve((size_t)N_NODES * 128 * 4);
    unsigned short* Afrag = (unsigned short*)carve((size_t)N_NODES * 256 * 2);  // reused L1/L2
    unsigned short* Wf1   = (unsigned short*)carve((size_t)2 * IN_CH  * HID_CH * 2);
    unsigned short* Wf2   = (unsigned short*)carve((size_t)2 * HID_CH * OUT_CH * 2);

    const int NT = N_NODES / 16;                       // 6250 node tiles (exact)
    const size_t APREP_N = (size_t)NT * 8 * 32 * 16;   // = N_NODES * 256

    // 0) zero cnt + agg
    k_fill0<<<2048, 256, 0, stream>>>(cnt, (size_t)N_NODES);
    k_fill0<<<8192, 256, 0, stream>>>(agg, (size_t)N_NODES * 128);

    // 1) degrees (shared by both layers)
    k_count<<<(N_EDGES + 255) / 256, 256, 0, stream>>>(dst, cnt, N_EDGES);

    // 2) pack weights -> bf16 fragment order
    k_packw<<<(2 * IN_CH * HID_CH + 255) / 256, 256, 0, stream>>>(Wl1, Wr1, Wf1, IN_CH, HID_CH);
    k_packw<<<(2 * HID_CH * OUT_CH + 255) / 256, 256, 0, stream>>>(Wl2, Wr2, Wf2, HID_CH, OUT_CH);

    // 3) layer 1: scatter x -> agg; build A fragments; WMMA GEMM -> h (ReLU)
    k_scatter<<<N_EDGES / 2, 256, 0, stream>>>(x, src, dst, agg, N_EDGES);
    k_aprep<<<(int)((APREP_N + 255) / 256), 256, 0, stream>>>(agg, x, cnt, Afrag, IN_CH);
    k_sage_gemm<IN_CH, HID_CH, 1><<<NT, 32 * (HID_CH / 16), 0, stream>>>(Afrag, Wf1, b1, h);

    // 4) layer 2: re-zero agg (buffer reuse keeps us L2-resident); scatter h;
    //    rebuild A fragments; WMMA GEMM -> out
    k_fill0<<<8192, 256, 0, stream>>>(agg, (size_t)N_NODES * 128);
    k_scatter<<<N_EDGES / 2, 256, 0, stream>>>(h, src, dst, agg, N_EDGES);
    k_aprep<<<(int)((APREP_N + 255) / 256), 256, 0, stream>>>(agg, h, cnt, Afrag, HID_CH);
    k_sage_gemm<HID_CH, OUT_CH, 0><<<NT, 32 * (OUT_CH / 16), 0, stream>>>(Afrag, Wf2, b2, out);
}
